// DynamicConv2d_86801289052697
// MI455X (gfx1250) — matrix operand order repaired and checked by reading, not compiled, not run on hardware
//
#include <hip/hip_runtime.h>

// ---------------- problem constants ----------------
#define BATCH    32
#define C_IN     64
#define HH       128
#define WW       128
#define C_OUT    64
#define KSZ      3
#define KK       4
#define LANG_DIM 768
#define HIDDEN   192
#define KDIM     (C_IN * KSZ * KSZ)     // 576
#define KSTEPS   (KDIM / 32)            // 18 WMMA K-steps of 32
// K ordering for the GEMM: k = (kh*3+kw)*64 + c   (tap-major, channel-minor)

typedef __attribute__((ext_vector_type(16))) __bf16        v16bf;
typedef __attribute__((ext_vector_type(8)))  float         v8f;
typedef __attribute__((ext_vector_type(4)))  int           vi4;
typedef __attribute__((ext_vector_type(4)))  unsigned int  u32x4;
typedef __attribute__((ext_vector_type(8)))  int           i32x8;
typedef __attribute__((ext_vector_type(4)))  int           i32x4;

union Frag16 {
    v16bf          v;
    __bf16         h[16];
    unsigned short u[16];
    vi4            q[2];
};

__device__ __forceinline__ unsigned short f2bf(float f) {
    union { float f; unsigned u; } cv; cv.f = f;
    unsigned u = cv.u;
    unsigned r = (u + 0x7FFFu + ((u >> 16) & 1u)) >> 16;   // round-to-nearest-even
    return (unsigned short)r;
}

// ---------------------------------------------------------------------------
// Kernel 1: attention head.  One block per sample, 192 threads.
// h = relu(lang @ fc1_w.T); logits = (h @ fc2_w.T + b)/31; softmax.
// ---------------------------------------------------------------------------
__global__ __launch_bounds__(HIDDEN)
void attn_kernel(const float* __restrict__ lang,
                 const float* __restrict__ fc1_w,
                 const float* __restrict__ fc2_w,
                 const float* __restrict__ fc2_b,
                 float* __restrict__ attn_ws) {
    __shared__ float h[HIDDEN];
    __shared__ float logits[KK];
    const int b = blockIdx.x;
    const int t = threadIdx.x;

    float acc = 0.f;
    const float* lrow = lang + (size_t)b * LANG_DIM;
    const float* wrow = fc1_w + (size_t)t * LANG_DIM;
    for (int l = 0; l < LANG_DIM; ++l) acc += lrow[l] * wrow[l];
    h[t] = acc > 0.f ? acc : 0.f;
    __syncthreads();

    if (t < KK) {
        float s = fc2_b[t];
        const float* w2 = fc2_w + (size_t)t * HIDDEN;
        for (int j = 0; j < HIDDEN; ++j) s += h[j] * w2[j];
        logits[t] = s * (1.0f / 31.0f);
    }
    __syncthreads();

    if (t == 0) {
        float m = logits[0];
        for (int k = 1; k < KK; ++k) m = fmaxf(m, logits[k]);
        float e[KK], sum = 0.f;
        for (int k = 0; k < KK; ++k) { e[k] = __expf(logits[k] - m); sum += e[k]; }
        float inv = 1.0f / sum;
        for (int k = 0; k < KK; ++k) attn_ws[b * KK + k] = e[k] * inv;
    }
}

// ---------------------------------------------------------------------------
// Kernel 2: per-sample bias aggregation.  bias[b][o] = attn[b] . cand_b[:,o]
// ---------------------------------------------------------------------------
__global__ __launch_bounds__(256)
void bias_kernel(const float* __restrict__ attn_ws,
                 const float* __restrict__ cand_b,
                 float* __restrict__ bias_ws) {
    int tid = blockIdx.x * blockDim.x + threadIdx.x;
    if (tid >= BATCH * C_OUT) return;
    int b = tid / C_OUT, o = tid % C_OUT;
    float s = 0.f;
    for (int j = 0; j < KK; ++j) s += attn_ws[b * KK + j] * cand_b[j * C_OUT + o];
    bias_ws[tid] = s;
}

// ---------------------------------------------------------------------------
// Kernel 3: aggregate candidate kernels with attn weights and pack into the
// bf16 A 16x32 fragment layout used here (contiguous k per lane half, same order as B):
//   k_local = e + (lane<16?0:16)
// GEMM K ordering: k = (kh*3+kw)*64 + c.
// Stored per (b, mtile, kstep) as 32 lanes x 16 bf16 = 1 KiB blocks so the
// conv kernel fetches a fragment with two contiguous b128 loads per lane.
// ---------------------------------------------------------------------------
__global__ __launch_bounds__(256)
void packA_kernel(const float* __restrict__ attn_ws,
                  const float* __restrict__ cand_w,
                  unsigned short* __restrict__ packedA) {
    const long long total = (long long)BATCH * 4 * KSTEPS * 32 * 16;
    long long g = (long long)blockIdx.x * blockDim.x + threadIdx.x;
    if (g >= total) return;

    int e    = (int)(g & 15);
    int lane = (int)((g >> 4) & 31);
    long long rest = g >> 9;
    int kt = (int)(rest % KSTEPS); rest /= KSTEPS;
    int mt = (int)(rest % 4);
    int b  = (int)(rest / 4);

    int k_local = e + (lane < 16 ? 0 : 16);
    int k = kt * 32 + k_local;                  // 0..575
    int o = mt * 16 + (lane & 15);
    int c = k & 63;                             // channel
    int q = k >> 6;                             // tap: kh*3+kw

    float s = 0.f;
    for (int j = 0; j < KK; ++j) {
        // cand_w[j][o][c][kh][kw]
        size_t idx = (((size_t)j * C_OUT + o) * C_IN + c) * (KSZ * KSZ) + q;
        s += attn_ws[b * KK + j] * cand_w[idx];
    }
    size_t off = (((size_t)(b * 4 + mt) * KSTEPS + kt) * 512) + (size_t)lane * 16 + e;
    packedA[off] = f2bf(s);
}

// ---------------------------------------------------------------------------
// Kernel 4: implicit-GEMM conv via v_wmma_f32_16x16x32_bf16.
// One block = (sample b, output row y); GEMM [64 x 576] * [576 x 128].
// The 3 contributing input rows (fp32, 64 ch each) are DMA'd into LDS by the
// Tensor Data Mover: one 2D tensor_load_to_lds per row, tensor_dim0=128,
// tile 128x64, tensor_dim0_stride = H*W (channel stride), with TDM LDS
// padding (128-dword lines + 10-dword pad) creating a 138-dword channel slot:
//   slot layout: [0..2] unused, [3] = x=-1 halo, [4..131] = x 0..127,
//                [132] = x=128 halo, [133..137] pad.
// 16*138 mod 64 = 32 -> the two lane halves of a B-fragment gather hit
// disjoint bank halves.  bf16 conversion happens in-register at gather time.
// ---------------------------------------------------------------------------
#define CSTRIDE 138                       // dwords per channel slot in LDS
__global__ __launch_bounds__(256)
void conv_kernel(const float* __restrict__ x,
                 const unsigned short* __restrict__ packedA,
                 const float* __restrict__ bias_ws,
                 float* __restrict__ out) {
    __shared__ float xlds[3 * C_IN * CSTRIDE];          // 105,984 B

    const int y = blockIdx.x;
    const int b = blockIdx.y;
    const int t = threadIdx.x;
    const int lane = t & 31;
    const int wave = t >> 5;

    // ---- wave 0: issue one TDM descriptor per valid input row ----
    if (wave == 0) {
        for (int kh = 0; kh < 3; ++kh) {
            int yg = y + kh - 1;
            if (0 <= yg && yg < HH) {
                unsigned lds_base =
                    (unsigned)(uintptr_t)&xlds[(kh * C_IN) * CSTRIDE + 4];
                unsigned long long ga = (unsigned long long)(uintptr_t)
                    (x + ((size_t)b * C_IN * HH + yg) * WW);
                u32x4 g0;
                g0[0] = 1u;                                   // count=1
                g0[1] = lds_base;                             // lds_addr (bytes)
                g0[2] = (unsigned)ga;                         // global_addr lo
                g0[3] = (unsigned)((ga >> 32) & 0x01FFFFFFu)  // global_addr hi
                        | 0x80000000u;                        // type=2 (image)
                i32x8 g1;
                g1[0] = (2 << 16)        // data_size = 4 bytes
                      | (1 << 20)        // pad_enable
                      | (6 << 22)        // pad_interval: 128 dwords
                      | (9 << 25);       // pad_amount: 10 dwords
                g1[1] = (WW << 16);      // tensor_dim0[15:0] = 128
                g1[2] = (C_IN << 16);    // dim0 hi=0 | tensor_dim1[15:0] = 64
                g1[3] = (WW << 16);      // dim1 hi=0 | tile_dim0 = 128
                g1[4] = C_IN;            // tile_dim1 = 64 | tile_dim2 = 0
                g1[5] = HH * WW;         // tensor_dim0_stride lo = 16384
                g1[6] = 0;               // stride hi | dim1_stride lo
                g1[7] = 0;
                i32x4 gz4 = {0, 0, 0, 0};
                i32x8 gz8 = {0, 0, 0, 0, 0, 0, 0, 0};
                __builtin_amdgcn_tensor_load_to_lds(g0, g1, gz4, gz4, gz8, 0);
            }
        }
    }

    // ---- all threads: zero halo columns (and whole slab for OOB rows) ----
    for (int idx = t; idx < 3 * C_IN; idx += 256) {
        float* slot = &xlds[idx * CSTRIDE];
        slot[3] = 0.f;                   // x = -1
        slot[132] = 0.f;                 // x = 128
    }
    #pragma unroll
    for (int kh = 0; kh < 3; ++kh) {
        int yg = y + kh - 1;
        if (yg < 0 || yg >= HH) {
            for (int idx = t; idx < C_IN * WW; idx += 256) {
                int c = idx >> 7, xg = idx & 127;
                xlds[(kh * C_IN + c) * CSTRIDE + 4 + xg] = 0.f;
            }
        }
    }

    if (wave == 0) __builtin_amdgcn_s_wait_tensorcnt(0);
    __syncthreads();

    // ---- main GEMM loop ----
    const int ncol = wave * 16 + (lane & 15);   // output x this lane feeds
    const int kb   = (lane < 16) ? 0 : 16;      // B layout lane-half K offset

    v8f acc[4];
    #pragma unroll
    for (int m = 0; m < 4; ++m) acc[m] = (v8f){0.f,0.f,0.f,0.f,0.f,0.f,0.f,0.f};

    const vi4* __restrict__ pAq =
        (const vi4*)(packedA + (size_t)b * 4 * KSTEPS * 512);

    for (int kt = 0; kt < KSTEPS; ++kt) {
        int q  = kt >> 1;                       // tap index (uniform -> SALU)
        int kh = q / 3;
        int kw = q - 3 * kh;
        int cb = ((kt & 1) << 5) + kb;          // channel base for this half
        // gather 16 channels at fixed (tap, column); convert fp32 -> bf16
        const float* src = &xlds[(kh * C_IN + cb) * CSTRIDE + 3 + ncol + kw];
        Frag16 bF;
        #pragma unroll
        for (int e = 0; e < 16; ++e)
            bF.h[e] = (__bf16)src[e * CSTRIDE];

        #pragma unroll
        for (int mt = 0; mt < 4; ++mt) {
            Frag16 aF;
            size_t qidx = ((size_t)(mt * KSTEPS + kt)) * 64 + (size_t)lane * 2;
            aF.q[0] = pAq[qidx];
            aF.q[1] = pAq[qidx + 1];
            acc[mt] = __builtin_amdgcn_wmma_f32_16x16x32_bf16(
                false, aF.v, false, bF.v, (short)0, acc[mt], false, false);
        }
    }

    // ---- epilogue: C/D layout (lane<16: M=r, lane>=16: M=r+8), add bias ----
    const int mofs = (lane < 16) ? 0 : 8;
    const int xcol = wave * 16 + (lane & 15);
    #pragma unroll
    for (int mt = 0; mt < 4; ++mt) {
        #pragma unroll
        for (int r = 0; r < 8; ++r) {
            int o = mt * 16 + r + mofs;
            out[(((size_t)b * C_OUT + o) * HH + y) * WW + xcol] =
                acc[mt][r] + bias_ws[b * C_OUT + o];
        }
    }
}

// ---------------------------------------------------------------------------
extern "C" void kernel_launch(void* const* d_in, const int* in_sizes, int n_in,
                              void* d_out, int out_size, void* d_ws, size_t ws_size,
                              hipStream_t stream) {
    const float* x      = (const float*)d_in[0];
    const float* lang   = (const float*)d_in[1];
    const float* fc1_w  = (const float*)d_in[2];
    const float* fc2_w  = (const float*)d_in[3];
    const float* fc2_b  = (const float*)d_in[4];
    const float* cand_w = (const float*)d_in[5];
    const float* cand_b = (const float*)d_in[6];
    float* out = (float*)d_out;

    // workspace layout
    float* attn_ws = (float*)d_ws;                        // 128 floats
    float* bias_ws = attn_ws + 128;                       // 2048 floats
    unsigned short* packedA = (unsigned short*)(bias_ws + 2048);  // 16B-aligned

    attn_kernel<<<BATCH, HIDDEN, 0, stream>>>(lang, fc1_w, fc2_w, fc2_b, attn_ws);

    bias_kernel<<<(BATCH * C_OUT + 255) / 256, 256, 0, stream>>>(attn_ws, cand_b, bias_ws);

    const long long packTotal = (long long)BATCH * 4 * KSTEPS * 32 * 16;   // 1,179,648
    packA_kernel<<<(int)((packTotal + 255) / 256), 256, 0, stream>>>(attn_ws, cand_w, packedA);

    dim3 grid(HH, BATCH);
    conv_kernel<<<grid, 256, 0, stream>>>(x, packedA, bias_ws, out);
}
